// CrossModalFeatureInteractionModule_24489903522351
// MI455X (gfx1250) — matrix operand order repaired and checked by reading, not compiled, not run on hardware
//
#include <hip/hip_runtime.h>

typedef __attribute__((ext_vector_type(16))) __bf16 v16bf;
typedef __attribute__((ext_vector_type(2)))  __bf16 v2bf;
typedef __attribute__((ext_vector_type(8)))  float  v8f;
typedef __attribute__((ext_vector_type(4)))  float  v4f;

#define C_     256
#define HW_    4096
#define NPIX   65536
#define NTILES 4096            // 16-pixel tiles
#define TPW    2               // tiles per wave
#define WPB    2               // waves per block
#define THREADS (WPB * 32)
#define NBLOCKS (NTILES / (TPW * WPB))   // 1024
#define SROW   528             // bf16 elements per LDS pixel-row (512 + 16 pad; 1056B, 16B-aligned)

__device__ __forceinline__ unsigned short f2bf(float x) {
    __bf16 h = (__bf16)x;
    return __builtin_bit_cast(unsigned short, h);
}

// ---- weight prep: fp32 -> bf16, pre-swizzled into WMMA A-matrix lane layout ----
// 16-bit A 16x32, this file's contiguous K order: lane L holds row M=L%16; element e -> K = (L<16?0:16) + e
__global__ void prep_weights_kernel(const float* __restrict__ W1,
                                    const float* __restrict__ W2,
                                    unsigned short* __restrict__ w1bf,   // [16 kb][2 mt][32 lane][16 e]
                                    unsigned short* __restrict__ w2bf) { // [16 t][32 lane][16 e]
    int i = blockIdx.x * blockDim.x + threadIdx.x;
    if (i < 16 * 2 * 32 * 16) {
        int e    = i & 15;
        int lane = (i >> 4) & 31;
        int mt   = (i >> 9) & 1;
        int kb   = i >> 10;
        int kidx = (lane & 16) + e;
        int row  = mt * 16 + (lane & 15);
        w1bf[i] = f2bf(W1[row * 512 + kb * 32 + kidx]);
    }
    int j = i - 16 * 2 * 32 * 16;
    if (j >= 0 && j < 16 * 32 * 16) {
        int e    = j & 15;
        int lane = (j >> 4) & 31;
        int t    = j >> 9;
        int kidx = (lane & 16) + e;
        w2bf[j] = f2bf(W2[(t * 16 + (lane & 15)) * 32 + kidx]);
    }
}

// ---- fused kernel ----
// GEMM1: D1[hid32 x px16] = W1[32x512] x X[512 x px16]  (bf16 WMMA, K=512 in 16 steps)
// GEMM2: D2[256  x px16] = W2[256x32] x relu(D1 + b1)   (16 M-tiles, K=32)
// out   = vis + ir + D2 + b2
__global__ __launch_bounds__(THREADS) void cmfi_fused_kernel(
    const float* __restrict__ vis, const float* __restrict__ ir,
    const float* __restrict__ b1,  const float* __restrict__ b2,
    const unsigned short* __restrict__ w1bf,
    const unsigned short* __restrict__ w2bf,
    float* __restrict__ out)
{
    // bf16 staging: row = pixel (16 rows), col = concat channel (512) + pad
    __shared__ unsigned short smem[WPB][16 * SROW];

    const int tid   = threadIdx.x;
    const int wave  = tid >> 5;
    const int lane  = tid & 31;
    const int pix   = lane & 15;
    const int hi    = (lane >> 4) & 1;
    const int half8 = hi * 8;
    unsigned short* sm = smem[wave];

    const int cidx = lane >> 2;       // 0..7: channel-pair index for staging loads
    const int ppx  = (lane & 3) * 4;  // pixel group start for staging loads

    const v16bf* w1v = reinterpret_cast<const v16bf*>(w1bf);
    const v16bf* w2v = reinterpret_cast<const v16bf*>(w2bf);

    // b1 biases for the D1 rows this lane owns
    float b1lo[8], b1hi[8];
#pragma unroll
    for (int r = 0; r < 8; ++r) {
        b1lo[r] = b1[half8 + r];
        b1hi[r] = b1[16 + half8 + r];
    }

    const int gwave = blockIdx.x * WPB + wave;

    for (int it = 0; it < TPW; ++it) {
        const int tile = gwave * TPW + it;
        const int p0   = tile * 16;
        const int bidx = p0 >> 12;
        const int sp   = p0 & (HW_ - 1);
        const size_t pbase = (size_t)bidx * C_ * HW_ + (size_t)sp;
        const float* visP = vis + pbase;     // pixel-tile base (no lane offset)
        const float* irP  = ir  + pbase;
        float*       outP = out + pbase;

        v8f t0 = {0.f,0.f,0.f,0.f,0.f,0.f,0.f,0.f};
        v8f t1 = {0.f,0.f,0.f,0.f,0.f,0.f,0.f,0.f};

        // ---- GEMM1: K = 512 in 16 steps of 32 ----
#pragma unroll 8
        for (int kb = 0; kb < 16; ++kb) {
            const float* src = (kb < 8) ? visP : irP;
            const int cm = (kb & 7) * 32;            // channel base within modality

            // stage 32 channels x 16 pixels: lane loads a channel PAIR (c, c+1) x 4 pixels,
            // packs bf16 pairs -> 8 ds_store_b32 per K-step (channel-contiguous dwords)
#pragma unroll
            for (int g = 0; g < 2; ++g) {
                const int cp = g * 16 + 2 * cidx;    // even channel offset within the 32-block
                const float* sp0 = src + (size_t)(cm + cp) * HW_ + ppx;
                v4f xa = *(const v4f*)(sp0);         // channel c,   4 consecutive pixels
                v4f xb = *(const v4f*)(sp0 + HW_);   // channel c+1, same pixels
                const int col = kb * 32 + cp;        // concat-channel (K) column
#pragma unroll
                for (int i = 0; i < 4; ++i) {
                    v2bf pr;
                    pr[0] = (__bf16)xa[i];
                    pr[1] = (__bf16)xb[i];           // v_cvt_pk_bf16_f32
                    *(unsigned int*)(sm + (ppx + i) * SROW + col) =
                        __builtin_bit_cast(unsigned int, pr);
                }
            }
            __builtin_amdgcn_wave_barrier();   // LDS stores (all lanes) before transposed read

            // B operand: one aligned 32B LDS read -> v16bf (half-split K layout)
            v16bf bv = *(const v16bf*)(sm + pix * SROW + kb * 32 + hi * 16);

            v16bf a0 = w1v[(kb * 2 + 0) * 32 + lane];   // W1 rows 0..15 (pre-swizzled bf16)
            v16bf a1 = w1v[(kb * 2 + 1) * 32 + lane];   // W1 rows 16..31
            t0 = __builtin_amdgcn_wmma_f32_16x16x32_bf16(false, a0, false, bv, (short)0, t0, false, false);
            t1 = __builtin_amdgcn_wmma_f32_16x16x32_bf16(false, a1, false, bv, (short)0, t1, false, false);
        }

        // ---- bias + relu, reshape D1 -> GEMM2 B operand ----
        float u0[8], u1[8];
#pragma unroll
        for (int r = 0; r < 8; ++r) {
            u0[r] = fmaxf(t0[r] + b1lo[r], 0.f);   // hid = half8 + r
            u1[r] = fmaxf(t1[r] + b1hi[r], 0.f);   // hid = 16 + half8 + r
        }
        float s0[8], s1[8];
#pragma unroll
        for (int r = 0; r < 8; ++r) {              // swap lane halves (permlanex16)
            s0[r] = __shfl_xor(u0[r], 16, 32);
            s1[r] = __shfl_xor(u1[r], 16, 32);
        }
        v16bf hv;                                   // lane<16: K=0..15, lane>=16: K=16..31
#pragma unroll
        for (int r = 0; r < 8; ++r) {
            hv[r]     = (__bf16)(hi ? s1[r] : u0[r]);
            hv[8 + r] = (__bf16)(hi ? u1[r] : s0[r]);
        }

        // ---- GEMM2 (K=32) + fused residual epilogue ----
#pragma unroll 2
        for (int t = 0; t < 16; ++t) {
            v16bf a2 = w2v[t * 32 + lane];
            v8f z = {0.f,0.f,0.f,0.f,0.f,0.f,0.f,0.f};
            v8f d = __builtin_amdgcn_wmma_f32_16x16x32_bf16(false, a2, false, hv, (short)0, z, false, false);

            const int c0 = t * 16 + half8;          // D rows this lane owns
            v4f bb0 = *(const v4f*)(b2 + c0);
            v4f bb1 = *(const v4f*)(b2 + c0 + 4);
#pragma unroll
            for (int r = 0; r < 8; ++r) {
                const int c = c0 + r;
                // vis/ir re-reads hit WGP$/L2 (same lines fetched by GEMM1 staging above)
                float vsum = visP[(size_t)c * HW_ + pix] + irP[(size_t)c * HW_ + pix];
                float bias = (r < 4) ? bb0[r] : bb1[r - 4];
                outP[(size_t)c * HW_ + pix] = vsum + bias + d[r];   // coalesced store
            }
        }
    }
}

extern "C" void kernel_launch(void* const* d_in, const int* in_sizes, int n_in,
                              void* d_out, int out_size, void* d_ws, size_t ws_size,
                              hipStream_t stream) {
    (void)in_sizes; (void)n_in; (void)out_size; (void)ws_size;
    const float* vis = (const float*)d_in[0];
    const float* ir  = (const float*)d_in[1];
    const float* W1  = (const float*)d_in[2];
    const float* b1  = (const float*)d_in[3];
    const float* W2  = (const float*)d_in[4];
    const float* b2  = (const float*)d_in[5];
    float* out = (float*)d_out;

    unsigned short* w1bf = (unsigned short*)d_ws;            // 16384 bf16 = 32 KB
    unsigned short* w2bf = w1bf + 16 * 2 * 32 * 16;          //  8192 bf16 = 16 KB

    const int prepN = 16 * 2 * 32 * 16 + 16 * 32 * 16;       // 24576
    prep_weights_kernel<<<(prepN + 255) / 256, 256, 0, stream>>>(W1, W2, w1bf, w2bf);
    cmfi_fused_kernel<<<NBLOCKS, THREADS, 0, stream>>>(vis, ir, b1, b2, w1bf, w2bf, out);
}
